// DeltaNetIVON_15178414424266
// MI455X (gfx1250) — compile-verified, matrix operand order checked by reading, not run on hardware
//
#include <hip/hip_runtime.h>

// ---------------- problem dims (fixed by reference) ----------------
#define B_    2
#define T_    1024
#define D_    512
#define H_    8
#define MT_   (B_ * T_)          // 2048 rows
#define C_    512                // H*K == H*V == 512 channels

#define LR_    1e-3f
#define BETA1_ 0.9f
#define BETA2_ 0.999f
#define EPS_   1e-5f
#define SCALE_ 0.125f            // 64^-0.5

typedef __attribute__((ext_vector_type(16))) _Float16 v16h;
typedef __attribute__((ext_vector_type(8)))  float    v8f;

// ---------------- fp32 -> f16 conversion (row-major, for A operands) -------------
__global__ void f32_to_f16(const float* __restrict__ in, _Float16* __restrict__ out, int n) {
    int i = blockIdx.x * blockDim.x + threadIdx.x;
    if (i < n) out[i] = (_Float16)in[i];
}

// ---------------- pack fp32 weight [Kd x N] into WMMA B-fragment layout ----------
// Bp layout: [nt = n/16][ks = k/32][lane 0..31][e 0..15]  (contiguous 32B per lane)
// Element map (ISA 7.12.2, 16-bit B 32x16): lane -> col = lane&15, kh = lane>>4;
// e -> k = ks*32 + (e<8 ? e : e+8) + 8*kh
__global__ void pack_b_f16(const float* __restrict__ W, _Float16* __restrict__ Bp,
                           int Kd, int N) {
    int idx = blockIdx.x * blockDim.x + threadIdx.x;     // over Kd*N
    if (idx >= Kd * N) return;
    const int e    = idx & 15;
    const int lane = (idx >> 4) & 31;
    const int ks   = (idx >> 9) % (Kd / 32);
    const int nt   = idx / ((Kd / 32) << 9);
    const int kh   = lane >> 4;
    const int col  = lane & 15;
    const int k    = ks * 32 + ((e < 8 ? e : e + 8) + 8 * kh);
    Bp[idx] = (_Float16)W[(size_t)k * N + nt * 16 + col];
}

// ---------------- WMMA f16 GEMM: C[MxN] = A[MxKd] * Bpacked ----------------------
// block = 128 threads (4 waves); each wave owns one 16x16 tile; k-loop fully
// unrolled (Kd/32 = 16 static v_wmma_f32_16x16x32_f16 per wave).
// grid = (N/64, M/16)
template <int Kd, int N>
__global__ __launch_bounds__(128)
void gemm_f16_wmma(const _Float16* __restrict__ A, const _Float16* __restrict__ Bp,
                   float* __restrict__ C) {
    const int lane = threadIdx.x & 31;
    const int wave = threadIdx.x >> 5;
    const int nt   = blockIdx.x * 4 + wave;  // 16-wide n-tile index
    const int m0   = blockIdx.y * 16;
    const int r    = lane & 15;              // A-row / C-col within tile
    const int kh   = lane >> 4;              // K-half select

    const _Float16* Arow  = A + (size_t)(m0 + r) * Kd + 8 * kh;
    const _Float16* Bfrag = Bp + ((size_t)nt * (Kd / 32) * 32 + lane) * 16;

    v8f acc = {};
#pragma unroll
    for (int ks = 0; ks < Kd / 32; ++ks) {
        v16h a;
        const _Float16* ap = Arow + ks * 32;
#pragma unroll
        for (int e = 0; e < 8; ++e) {        // k = ks*32 + 8*kh + e (+16 for hi half)
            a[e]     = ap[e];
            a[e + 8] = ap[e + 16];
        }
        const v16h b = *(const v16h*)(Bfrag + (size_t)ks * 32 * 16);
        acc = __builtin_amdgcn_wmma_f32_16x16x32_f16(false, a, false, b,
                                                     (short)0, acc, false, false);
    }
    // C/D layout: VGPR i -> row = i + 8*kh, col = lane&15
#pragma unroll
    for (int i = 0; i < 8; ++i)
        C[(size_t)(m0 + i + 8 * kh) * N + nt * 16 + r] = acc[i];
}

// ---------------- beta = sigmoid(x @ Wb), one thread per (row, h) ----------------
__global__ void beta_sigmoid(const float* __restrict__ x, const float* __restrict__ Wb,
                             float* __restrict__ beta) {
    int idx = blockIdx.x * blockDim.x + threadIdx.x;   // over MT_*H_
    if (idx >= MT_ * H_) return;
    const int h = idx % H_;
    const int row = idx / H_;
    float acc = 0.f;
    for (int d = 0; d < D_; ++d)
        acc += x[(size_t)row * D_ + d] * Wb[(size_t)d * H_ + h];
    beta[idx] = __builtin_amdgcn_rcpf(1.f + __expf(-acc));
}

// ---------------- causal depthwise conv-4 + SiLU, one thread per element --------
__global__ void conv_silu(const float* __restrict__ P, const float* __restrict__ w,
                          float* __restrict__ Y) {
    int idx = blockIdx.x * blockDim.x + threadIdx.x;   // over MT_*C_
    if (idx >= MT_ * C_) return;
    const int c  = idx % C_;
    const int bt = idx / C_;
    const int t  = bt % T_;
    float acc = 0.f;
#pragma unroll
    for (int i = 0; i < 4; ++i) {
        const int tt = t - 3 + i;
        if (tt >= 0) acc += P[(size_t)(bt - (3 - i)) * C_ + c] * w[c * 4 + i];
    }
    Y[idx] = acc * __builtin_amdgcn_rcpf(1.f + __expf(-acc));   // SiLU
}

// ---------------- per-head l2norm (sum, +1e-6) with scale (in-place safe) --------
__global__ __launch_bounds__(64)
void l2norm_scale(const float* __restrict__ in, float* __restrict__ out, float scale) {
    const int g = blockIdx.x;                 // over MT_*H_ groups of 64
    const int tid = threadIdx.x;
    const size_t base = (size_t)g * 64;
    const float x = in[base + tid];
    __shared__ float red[64];
    red[tid] = x * x;
    __syncthreads();
    for (int s = 32; s > 0; s >>= 1) {
        if (tid < s) red[tid] += red[tid + s];
        __syncthreads();
    }
    out[base + tid] = x * rsqrtf(red[0] + 1e-6f) * scale;
}

// ---------------- sequential IVON delta-rule recurrence --------------------------
// grid = (4 v-groups, H, B); block = 64 threads (kg = tid&3, vl = tid>>2).
// Double-buffered LDS (one barrier/step) + register prefetch of step t+1 inputs.
__global__ __launch_bounds__(64)
void recurrence(const float* __restrict__ qn, const float* __restrict__ kn,
                const float* __restrict__ vn, const float* __restrict__ beta,
                float* __restrict__ o) {
    const int tid = threadIdx.x;
    const int kg  = tid & 3;
    const int vl  = tid >> 2;
    const int vg  = blockIdx.x;
    const int h   = blockIdx.y;
    const int b   = blockIdx.z;
    const int v   = vg * 16 + vl;

    float S[16], G[16], Hm[16];
#pragma unroll
    for (int i = 0; i < 16; ++i) { S[i] = 0.f; G[i] = 0.f; Hm[i] = 0.f; }

    __shared__ float ks_[2][64], qs_[2][64], vs_[2][16];
    __shared__ float bsh[2];

    const size_t base  = (size_t)(b * T_) * C_ + h * 64;
    const size_t bbase = (size_t)(b * T_) * H_ + h;

    // preload step 0 into registers
    float kr = kn[base + tid];
    float qr = qn[base + tid];
    float vr = (tid < 16) ? vn[base + vg * 16 + tid] : 0.f;
    float br = beta[bbase];

    for (int t = 0; t < T_; ++t) {
        const int p = t & 1;
        ks_[p][tid] = kr;
        qs_[p][tid] = qr;
        if (tid < 16) vs_[p][tid] = vr;
        if (tid == 0) bsh[p] = br;

        if (t + 1 < T_) {                       // prefetch next step (overlaps compute)
            const size_t nb = base + (size_t)(t + 1) * C_;
            kr = kn[nb + tid];
            qr = qn[nb + tid];
            if (tid < 16) vr = vn[nb + vg * 16 + tid];
            br = beta[bbase + (size_t)(t + 1) * H_];
        }
        __syncthreads();

        // pred[v] = k . S[:,v]  (16 partials, reduced across the 4 lanes of a v)
        float pred = 0.f;
#pragma unroll
        for (int i = 0; i < 16; ++i) pred += ks_[p][kg * 16 + i] * S[i];
        pred += __shfl_xor(pred, 1, 32);
        pred += __shfl_xor(pred, 2, 32);

        const float err = pred - vs_[p][vl];
        const float lrb = LR_ * bsh[p];
        float og = 0.f;
#pragma unroll
        for (int i = 0; i < 16; ++i) {
            const float grad = ks_[p][kg * 16 + i] * err;   // outer(k, pred-v)
            G[i]  = BETA1_ * G[i]  + (1.f - BETA1_) * grad;
            Hm[i] = BETA2_ * Hm[i] + (1.f - BETA2_) * grad * grad;
            S[i] -= lrb * G[i] * __builtin_amdgcn_rcpf(Hm[i] + 1.f);  // WD==0, LAM==1
            og   += qs_[p][kg * 16 + i] * S[i];             // o[v] = q . S[:,v]
        }
        og += __shfl_xor(og, 1, 32);
        og += __shfl_xor(og, 2, 32);
        if (kg == 0) o[base + (size_t)t * C_ + v] = og;
    }
}

// ---------------- RMSNorm over V (mean, +1e-5) * rms_w, emit f16 -----------------
__global__ __launch_bounds__(64)
void rmsnorm_f16(const float* __restrict__ o, const float* __restrict__ rw,
                 _Float16* __restrict__ oh) {
    const int g = blockIdx.x;                 // over MT_*H_ groups of 64
    const int tid = threadIdx.x;
    const size_t base = (size_t)g * 64;
    const float x = o[base + tid];
    __shared__ float red[64];
    red[tid] = x * x;
    __syncthreads();
    for (int s = 32; s > 0; s >>= 1) {
        if (tid < s) red[tid] += red[tid + s];
        __syncthreads();
    }
    const float inv = rsqrtf(red[0] * (1.0f / 64.f) + EPS_);
    oh[base + tid] = (_Float16)(x * inv * rw[tid]);
}

// ---------------- host-side launch ----------------
extern "C" void kernel_launch(void* const* d_in, const int* in_sizes, int n_in,
                              void* d_out, int out_size, void* d_ws, size_t ws_size,
                              hipStream_t stream) {
    (void)in_sizes; (void)n_in; (void)out_size; (void)ws_size;
    const float* x    = (const float*)d_in[0];
    const float* Wq   = (const float*)d_in[1];
    const float* Wk   = (const float*)d_in[2];
    const float* Wv   = (const float*)d_in[3];
    const float* Wb   = (const float*)d_in[4];
    const float* cwq  = (const float*)d_in[5];
    const float* cwk  = (const float*)d_in[6];
    const float* cwv  = (const float*)d_in[7];
    const float* rmsw = (const float*)d_in[8];
    const float* Wo   = (const float*)d_in[9];
    float* out = (float*)d_out;

    char* ws = (char*)d_ws;
    size_t off = 0;
    auto alloc = [&](size_t bytes) -> void* {
        void* p = ws + off;
        off += (bytes + 255) & ~(size_t)255;
        return p;
    };
    _Float16* xh   = (_Float16*)alloc((size_t)MT_ * D_ * 2);
    _Float16* Wqp  = (_Float16*)alloc((size_t)D_ * C_ * 2);   // packed fragments
    _Float16* Wkp  = (_Float16*)alloc((size_t)D_ * C_ * 2);
    _Float16* Wvp  = (_Float16*)alloc((size_t)D_ * C_ * 2);
    _Float16* Wop  = (_Float16*)alloc((size_t)C_ * D_ * 2);
    float*    Qf   = (float*)alloc((size_t)MT_ * C_ * 4);
    float*    Kf   = (float*)alloc((size_t)MT_ * C_ * 4);
    float*    Vf   = (float*)alloc((size_t)MT_ * C_ * 4);
    float*    qc   = (float*)alloc((size_t)MT_ * C_ * 4);
    float*    kc   = (float*)alloc((size_t)MT_ * C_ * 4);
    float*    vc   = (float*)alloc((size_t)MT_ * C_ * 4);
    float*    betaf= (float*)alloc((size_t)MT_ * H_ * 4);
    float*    ob   = (float*)alloc((size_t)MT_ * C_ * 4);
    _Float16* ohh  = (_Float16*)alloc((size_t)MT_ * C_ * 2);

    // 1) stage A operand (x) as f16; pack all weights into WMMA B-fragment layout
    int n = MT_ * D_;
    f32_to_f16<<<(n + 255) / 256, 256, 0, stream>>>(x, xh, n);
    n = D_ * C_;
    pack_b_f16<<<(n + 255) / 256, 256, 0, stream>>>(Wq, Wqp, D_, C_);
    pack_b_f16<<<(n + 255) / 256, 256, 0, stream>>>(Wk, Wkp, D_, C_);
    pack_b_f16<<<(n + 255) / 256, 256, 0, stream>>>(Wv, Wvp, D_, C_);
    pack_b_f16<<<(n + 255) / 256, 256, 0, stream>>>(Wo, Wop, C_, D_);

    // 2) projection GEMMs via v_wmma_f32_16x16x32_f16 (fully unrolled k-loop)
    dim3 gblk(128);
    dim3 ggrd(C_ / 64, MT_ / 16);
    gemm_f16_wmma<D_, C_><<<ggrd, gblk, 0, stream>>>(xh, Wqp, Qf);
    gemm_f16_wmma<D_, C_><<<ggrd, gblk, 0, stream>>>(xh, Wkp, Kf);
    gemm_f16_wmma<D_, C_><<<ggrd, gblk, 0, stream>>>(xh, Wvp, Vf);

    // 3) beta projection (tiny)
    n = MT_ * H_;
    beta_sigmoid<<<(n + 255) / 256, 256, 0, stream>>>(x, Wb, betaf);

    // 4) causal depthwise conv + SiLU
    n = MT_ * C_;
    conv_silu<<<(n + 255) / 256, 256, 0, stream>>>(Qf, cwq, qc);
    conv_silu<<<(n + 255) / 256, 256, 0, stream>>>(Kf, cwk, kc);
    conv_silu<<<(n + 255) / 256, 256, 0, stream>>>(Vf, cwv, vc);

    // 5) l2norm (in-place), scale on q
    l2norm_scale<<<MT_ * H_, 64, 0, stream>>>(qc, qc, SCALE_);
    l2norm_scale<<<MT_ * H_, 64, 0, stream>>>(kc, kc, 1.0f);

    // 6) sequential recurrence: 64 blocks (B*H*4 v-groups) x 64 threads
    dim3 rgrid(4, H_, B_);
    recurrence<<<rgrid, 64, 0, stream>>>(qc, kc, vc, betaf, ob);

    // 7) RMSNorm -> f16
    rmsnorm_f16<<<MT_ * H_, 64, 0, stream>>>(ob, rmsw, ohh);

    // 8) output projection GEMM (fp32 result straight to d_out)
    gemm_f16_wmma<C_, D_><<<ggrd, gblk, 0, stream>>>(ohh, Wop, out);
}
